// GraphPropagationBlock_78563541778924
// MI455X (gfx1250) — matrix operand order repaired and checked by reading, not compiled, not run on hardware
//
#include <hip/hip_runtime.h>
#include <hip/hip_bf16.h>
#include <math.h>

// ---------------------------------------------------------------------------
// GraphPropagationBlock for MI455X (gfx1250, wave32, WMMA bf16 path)
//
// Heavy GEMMs (QKV, q@k^T logits, attn-proj, fc1, fc2) run on
// v_wmma_f32_16x16x32_bf16. Main GEMM uses 2x2 register blocking
// (32x32 output per wave, 4 accumulators, A/B fragments reused twice)
// to double arithmetic intensity vs one-tile-per-wave.
// Irregular graph stages (softmax, pagerank in 154KB LDS, rank-select,
// bisection threshold, sparse propagate) are small VALU kernels.
// ---------------------------------------------------------------------------

typedef __attribute__((ext_vector_type(16))) __bf16       bf16x16;
typedef __attribute__((ext_vector_type(8)))  float        f32x8;
typedef __attribute__((ext_vector_type(4)))  unsigned int u32x4;

typedef unsigned short ushort_t;
typedef unsigned int   uint_t;

#define NB   64
#define NT   197
#define CH   768
#define NH   12
#define HD   64
#define NKP  99            // kept tokens (cls + 98)
#define NEL  98            // eliminated tokens
#define HID  3072
#define ROWS (NB*NT)       // 12608
#define RK   (NB*NKP)      // 6336
#define SA2  (NT*NT)       // 38809
#define NW   (NKP*NEL)     // 9702
#define KTH  1940          // int(NW*0.2)

union ABu { bf16x16 v; u32x4 q[2]; };

__device__ __forceinline__ ushort_t f2bf(float f) {
  uint_t u = __float_as_uint(f);
  u += 0x7FFFu + ((u >> 16) & 1u);
  return (ushort_t)(u >> 16);
}
__device__ __forceinline__ float bf2f(ushort_t h) {
  return __uint_as_float(((uint_t)h) << 16);
}

__device__ __forceinline__ float blockReduceSum(float v, float* red) {
  int tid = threadIdx.x;
  red[tid] = v; __syncthreads();
  for (int s = blockDim.x >> 1; s > 0; s >>= 1) {
    if (tid < s) red[tid] += red[tid + s];
    __syncthreads();
  }
  float r = red[0]; __syncthreads();
  return r;
}
__device__ __forceinline__ float blockReduceMax(float v, float* red) {
  int tid = threadIdx.x;
  red[tid] = v; __syncthreads();
  for (int s = blockDim.x >> 1; s > 0; s >>= 1) {
    if (tid < s) red[tid] = fmaxf(red[tid], red[tid + s]);
    __syncthreads();
  }
  float r = red[0]; __syncthreads();
  return r;
}

// --------------------------------------------------------------------------
// fp32 -> bf16 elementwise convert (weights, one-time per launch)
// --------------------------------------------------------------------------
__global__ void cvt_bf16_kernel(const float* __restrict__ in,
                                ushort_t* __restrict__ out, long n) {
  long i = (long)blockIdx.x * blockDim.x + threadIdx.x;
  if (i < n) out[i] = f2bf(in[i]);
}

// log(token_scales) precompute: removes v_log_f32 from the logits epilogue
__global__ void log_kernel(const float* __restrict__ in,
                           float* __restrict__ out, long n) {
  long i = (long)blockIdx.x * blockDim.x + threadIdx.x;
  if (i < n) out[i] = logf(in[i]);
}

// --------------------------------------------------------------------------
// LayerNorm over C=768, bf16 output (feeds WMMA GEMMs)
// --------------------------------------------------------------------------
__global__ void ln_bf16_kernel(const float* __restrict__ x,
                               const float* __restrict__ g,
                               const float* __restrict__ b,
                               ushort_t* __restrict__ out, int C) {
  __shared__ float red[256];
  long row = blockIdx.x;
  const float* xr = x + row * C;
  float s = 0.f;
  for (int c = threadIdx.x; c < C; c += 256) s += xr[c];
  float mu = blockReduceSum(s, red) / (float)C;
  float v = 0.f;
  for (int c = threadIdx.x; c < C; c += 256) { float d = xr[c] - mu; v += d * d; }
  float var = blockReduceSum(v, red) / (float)C;
  float rstd = rsqrtf(var + 1e-5f);
  for (int c = threadIdx.x; c < C; c += 256)
    out[row * C + c] = f2bf((xr[c] - mu) * rstd * g[c] + b[c]);
}

// --------------------------------------------------------------------------
// WMMA bf16 GEMM, 2x2 register-blocked: out[M,N] = A[M,K] @ W[N,K]^T
// One wave owns a 32x32 output block (4x 16x16 WMMA tiles); each A/B
// fragment is reused twice -> AI ~16 flop/byte.
//   EPI 0: out bf16, no bias                     (QKV)
//   EPI 1: out f32 = acc + bias[col] + resid     (proj, fc2)
//   EPI 2: out bf16 = GELU(acc + bias[col])      (fc1)
// M,N multiples of 32; K multiple of 32.
// --------------------------------------------------------------------------
template <int EPI>
__global__ void gemm_wmma_kernel(const ushort_t* __restrict__ A,
                                 const ushort_t* __restrict__ W,
                                 const float* __restrict__ bias,
                                 const float* __restrict__ resid,
                                 float* __restrict__ outF,
                                 ushort_t* __restrict__ outB,
                                 int M, int N, int K) {
  int wave = threadIdx.x >> 5;
  int lane = threadIdx.x & 31;
  int tn2 = N >> 5;
  long total = (long)(M >> 5) * tn2;
  long tile = (long)blockIdx.x * (blockDim.x >> 5) + wave;
  if (tile >= total) return;                        // wave-uniform exit
  int tm = (int)(tile / tn2), tn = (int)(tile % tn2);
  int l16 = lane & 15, lhi = lane >> 4;

  const ushort_t* A0 = A + (long)(tm * 32 + l16) * K;
  const ushort_t* A1 = A0 + (long)16 * K;
  const ushort_t* W0 = W + (long)(tn * 32 + l16) * K;
  const ushort_t* W1 = W0 + (long)16 * K;

  f32x8 c00 = {0.f,0.f,0.f,0.f,0.f,0.f,0.f,0.f};
  f32x8 c01 = c00, c10 = c00, c11 = c00;
  for (int k = 0; k < K; k += 32) {
    ABu a0, a1, b0, b1;
    a0.q[0] = *(const u32x4*)(const void*)(A0 + k + lhi * 8);       // K 0..7 / 8..15
    a0.q[1] = *(const u32x4*)(const void*)(A0 + k + 16 + lhi * 8);  // K 16..23 / 24..31
    a1.q[0] = *(const u32x4*)(const void*)(A1 + k + lhi * 8);
    a1.q[1] = *(const u32x4*)(const void*)(A1 + k + 16 + lhi * 8);
    b0.q[0] = *(const u32x4*)(const void*)(W0 + k + lhi * 8);       // K 0..7 / 8..15
    b0.q[1] = *(const u32x4*)(const void*)(W0 + k + 16 + lhi * 8);  // K 16..23 / 24..31
    b1.q[0] = *(const u32x4*)(const void*)(W1 + k + lhi * 8);
    b1.q[1] = *(const u32x4*)(const void*)(W1 + k + 16 + lhi * 8);
    c00 = __builtin_amdgcn_wmma_f32_16x16x32_bf16(false, a0.v, false, b0.v, (short)0, c00, false, false);
    c01 = __builtin_amdgcn_wmma_f32_16x16x32_bf16(false, a0.v, false, b1.v, (short)0, c01, false, false);
    c10 = __builtin_amdgcn_wmma_f32_16x16x32_bf16(false, a1.v, false, b0.v, (short)0, c10, false, false);
    c11 = __builtin_amdgcn_wmma_f32_16x16x32_bf16(false, a1.v, false, b1.v, (short)0, c11, false, false);
  }

  // epilogue: 4 sub-tiles
  for (int t = 0; t < 4; ++t) {
    f32x8 acc = (t == 0) ? c00 : (t == 1) ? c01 : (t == 2) ? c10 : c11;
    int row0 = tm * 32 + ((t >> 1) ? 16 : 0);
    int col  = tn * 32 + ((t & 1) ? 16 : 0) + l16;
    for (int r = 0; r < 8; ++r) {
      int row = row0 + r + 8 * lhi;                 // C/D layout: VGPRr -> M=r / r+8
      float vv = acc[r];
      long o = (long)row * N + col;
      if (EPI == 0) {
        outB[o] = f2bf(vv);
      } else if (EPI == 1) {
        outF[o] = vv + bias[col] + resid[o];
      } else {
        float tt = vv + bias[col];
        tt = 0.5f * tt * (1.f + erff(tt * 0.70710678118f));   // exact GELU
        outB[o] = f2bf(tt);
      }
    }
  }
}

// --------------------------------------------------------------------------
// Attention logits via WMMA: logits[b,h,i,j] = 0.125*(q_i . k_j) + lts[b,j]
// qkvb layout: [B,NT, 3*CH] bf16 (q at +0, k at +768, v at +1536; col=h*64+d)
// Grid: (ceil(169/4), NH, NB), block 128 (4 waves), 13x13 tiles of 16.
// --------------------------------------------------------------------------
__global__ void logits_wmma_kernel(const ushort_t* __restrict__ qkvb,
                                   const float* __restrict__ lts,
                                   float* __restrict__ attn) {
  int wave = threadIdx.x >> 5;
  int lane = threadIdx.x & 31;
  int tile = blockIdx.x * 4 + wave;
  if (tile >= 13 * 13) return;                      // wave-uniform exit
  int tm = tile / 13, tn = tile % 13;
  int b = blockIdx.z, h = blockIdx.y;
  int l16 = lane & 15, lhi = lane >> 4;

  int rowA = tm * 16 + l16; if (rowA > NT - 1) rowA = NT - 1;
  int rowB = tn * 16 + l16; if (rowB > NT - 1) rowB = NT - 1;
  const ushort_t* qp = qkvb + ((long)(b * NT + rowA) * (3 * CH) + h * HD);
  const ushort_t* kp = qkvb + ((long)(b * NT + rowB) * (3 * CH) + CH + h * HD);

  f32x8 acc = {0.f,0.f,0.f,0.f,0.f,0.f,0.f,0.f};
  for (int k = 0; k < HD; k += 32) {
    ABu a, bb;
    a.q[0]  = *(const u32x4*)(const void*)(qp + k + lhi * 8);
    a.q[1]  = *(const u32x4*)(const void*)(qp + k + 16 + lhi * 8);
    bb.q[0] = *(const u32x4*)(const void*)(kp + k + lhi * 8);
    bb.q[1] = *(const u32x4*)(const void*)(kp + k + 16 + lhi * 8);
    acc = __builtin_amdgcn_wmma_f32_16x16x32_bf16(
        false, a.v, false, bb.v, (short)0, acc, false, false);
  }
  int col = tn * 16 + l16;
  float bias = (col < NT) ? lts[b * NT + col] : 0.f;
  float* arow0 = attn + (long)(b * NH + h) * SA2;
  for (int r = 0; r < 8; ++r) {
    int row = tm * 16 + r + 8 * lhi;
    if (row < NT && col < NT)
      arow0[(long)row * NT + col] = acc[r] * 0.125f + bias;
  }
}

// --------------------------------------------------------------------------
// Row softmax in place on attn [B*H*NT rows of NT]
// --------------------------------------------------------------------------
__global__ void softmax_kernel(float* __restrict__ attn) {
  __shared__ float red[256];
  float* row = attn + (long)blockIdx.x * NT;
  float m = -1e30f;
  for (int j = threadIdx.x; j < NT; j += 256) m = fmaxf(m, row[j]);
  m = blockReduceMax(m, red);
  float s = 0.f;
  for (int j = threadIdx.x; j < NT; j += 256) {
    float e = __expf(row[j] - m);
    row[j] = e; s += e;
  }
  s = blockReduceSum(s, red);
  float inv = 1.f / s;
  for (int j = threadIdx.x; j < NT; j += 256) row[j] *= inv;
}

// --------------------------------------------------------------------------
// Head-average: avg[b,i,j] = mean_h attn[b,h,i,j]
// --------------------------------------------------------------------------
__global__ void attn_avg_kernel(const float* __restrict__ attn,
                                float* __restrict__ avg) {
  int b = blockIdx.x / NT, i = blockIdx.x % NT;
  for (int j = threadIdx.x; j < NT; j += 256) {
    float s = 0.f;
    for (int h = 0; h < NH; ++h)
      s += attn[((long)(b * NH + h) * NT + i) * NT + j];
    avg[((long)b * NT + i) * NT + j] = s * (1.f / NH);
  }
}

// --------------------------------------------------------------------------
// out[b,n,c] = sum_e attn[b,h(c),n,e] * v[b,e,c]   (bf16 out, feeds proj GEMM)
// All 12 attention rows for token n staged in LDS (9.3 KB) once per block.
// --------------------------------------------------------------------------
__global__ void attn_v_kernel(const float* __restrict__ attn,
                              const ushort_t* __restrict__ qkvb,
                              ushort_t* __restrict__ aob) {
  __shared__ float arows[NH * NT];   // 2364 floats
  int b = blockIdx.x / NT, n = blockIdx.x % NT;
  for (int idx = threadIdx.x; idx < NH * NT; idx += CH) {
    int h = idx / NT, e = idx % NT;
    arows[idx] = attn[((long)(b * NH + h) * NT + n) * NT + e];
  }
  __syncthreads();
  int c = threadIdx.x;               // 0..767
  int h = c >> 6;
  const float* ar = arows + h * NT;
  float acc = 0.f;
  for (int e = 0; e < NT; ++e)
    acc += ar[e] * bf2f(qkvb[(long)(b * NT + e) * (3 * CH) + 2 * CH + c]);
  aob[(long)(b * NT + n) * CH + c] = f2bf(acc);
}

// --------------------------------------------------------------------------
// PageRank (196x196 transition matrix in LDS) + rank-based kept/elim select.
// One block per batch b; dynamic LDS = ~159 KB (fits 320 KB/WGP).
// --------------------------------------------------------------------------
__global__ void pagerank_select_kernel(const float* __restrict__ avg,
                                       int* __restrict__ kept,
                                       int* __restrict__ elim) {
  extern __shared__ float sm[];
  const int N1 = NT - 1;             // 196
  float* Wl   = sm;                  // [196][197] padded
  float* pr   = Wl + N1 * (N1 + 1);
  float* prw  = pr + N1;
  float* rsum = prw + N1;
  float* npr  = rsum + N1;
  float* red  = npr + N1;            // 256
  int*   rank = (int*)(red + 256);   // 196
  int b = blockIdx.x, tid = threadIdx.x;
  const float* ab = avg + (long)b * SA2;

  for (int idx = tid; idx < N1 * N1; idx += 256) {
    int i = idx / N1, j = idx % N1;
    Wl[i * (N1 + 1) + j] = ab[(long)(i + 1) * NT + (j + 1)];
  }
  __syncthreads();
  for (int i = tid; i < N1; i += 256) {
    float s = 0.f;
    for (int j = 0; j < N1; ++j) s += Wl[i * (N1 + 1) + j];
    rsum[i] = s;
    pr[i] = 1.f / (float)N1;
  }
  __syncthreads();

  for (int it = 0; it < 20; ++it) {
    for (int j = tid; j < N1; j += 256) prw[j] = pr[j] / rsum[j];
    __syncthreads();
    float d2 = 0.f;
    for (int i = tid; i < N1; i += 256) {
      float s = 0.f;
      for (int j = 0; j < N1; ++j) s += Wl[j * (N1 + 1) + i] * prw[j];
      float nv = 0.95f * s + 0.05f / (float)N1;
      npr[i] = nv;
      float dd = nv - pr[i];
      d2 += dd * dd;
    }
    float dist2 = blockReduceSum(d2, red);
    for (int i = tid; i < N1; i += 256) pr[i] = npr[i];
    __syncthreads();
    if (dist2 < 1e-3f * 1e-3f) break;              // uniform
  }

  // stable descending rank (argsort(-pr), ties by ascending index)
  for (int i = tid; i < N1; i += 256) {
    float pi = pr[i];
    int r = 0;
    for (int k = 0; k < N1; ++k) {
      float pk = pr[k];
      r += (pk > pi) || (pk == pi && k < i);
    }
    rank[i] = r;
  }
  __syncthreads();
  if (tid == 0) {
    int* kb = kept + b * NKP;
    int* eb = elim + b * NEL;
    kb[0] = 0;
    int nk = 1, ne = 0;
    for (int i = 0; i < N1; ++i) {
      if (rank[i] < N1 - NEL) kb[nk++] = i + 1;    // ascending-sorted by build
      else                    eb[ne++] = i + 1;
    }
  }
}

// --------------------------------------------------------------------------
// Gather w[b,h,kept,elim] into LDS, find ~kth-largest threshold by value
// bisection (attn in [0,1]), write sparsified w to global.
// Grid (NH, NB), block 256, dynamic LDS ~= 40 KB.
// --------------------------------------------------------------------------
__global__ void gather_threshold_kernel(const float* __restrict__ attn,
                                        const int* __restrict__ kept,
                                        const int* __restrict__ elim,
                                        float* __restrict__ wg) {
  extern __shared__ float sm2[];
  float* wl = sm2;                   // NW
  int* ired = (int*)(wl + NW);       // 256
  int h = blockIdx.x, b = blockIdx.y, tid = threadIdx.x;
  const int* kb = kept + b * NKP;
  const int* eb = elim + b * NEL;
  const float* at = attn + (long)(b * NH + h) * SA2;
  float* wo = wg + (long)(b * NH + h) * NW;

  for (int idx = tid; idx < NW; idx += 256) {
    int ki = idx / NEL, ej = idx % NEL;
    wl[idx] = at[(long)kb[ki] * NT + eb[ej]];
  }
  __syncthreads();

  float lo = 0.f, hi = 1.f;
  for (int it = 0; it < 24; ++it) {
    float mid = 0.5f * (lo + hi);
    int c = 0;
    for (int idx = tid; idx < NW; idx += 256) c += (wl[idx] >= mid);
    ired[tid] = c; __syncthreads();
    for (int s = 128; s > 0; s >>= 1) {
      if (tid < s) ired[tid] += ired[tid + s];
      __syncthreads();
    }
    int tot = ired[0]; __syncthreads();
    if (tot > KTH) lo = mid; else hi = mid;        // keep count(>=lo) >= KTH+1
  }
  float thr = lo;
  for (int idx = tid; idx < NW; idx += 256) {
    float v = wl[idx];
    wo[idx] = (v >= thr) ? v : 0.f;
  }
}

// --------------------------------------------------------------------------
// ts_new[b,k] = ts[kept[k]] + sum_e mean_h(w[b,h,k,e]) * ts[elim[e]]
// --------------------------------------------------------------------------
__global__ void ts_new_kernel(const float* __restrict__ wg,
                              const float* __restrict__ ts,
                              const int* __restrict__ kept,
                              const int* __restrict__ elim,
                              float* __restrict__ tsout) {
  int b = blockIdx.x, k = threadIdx.x;
  if (k >= NKP) return;
  float acc = ts[b * NT + kept[b * NKP + k]];
  for (int e = 0; e < NEL; ++e) {
    float wm = 0.f;
    for (int h = 0; h < NH; ++h)
      wm += wg[((long)(b * NH + h) * NKP + k) * NEL + e];
    acc += (wm / (float)NH) * ts[b * NT + elim[b * NEL + e]];
  }
  tsout[b * NKP + k] = acc;
}

// --------------------------------------------------------------------------
// x_kept_final[b,k,c] = (ts_kept*x[kept] + sum_e w[b,h(c),k,e]*ts_e*x[elim_e]) / ts_new
// Grid (NKP, NB), block 768.
// --------------------------------------------------------------------------
__global__ void propagate_kernel(const float* __restrict__ xres,
                                 const float* __restrict__ wg,
                                 const float* __restrict__ ts,
                                 const float* __restrict__ tsnew,
                                 const int* __restrict__ kept,
                                 const int* __restrict__ elim,
                                 float* __restrict__ xk) {
  int k = blockIdx.x, b = blockIdx.y, c = threadIdx.x;
  int h = c >> 6;
  int kt = kept[b * NKP + k];
  float acc = xres[((long)b * NT + kt) * CH + c] * ts[b * NT + kt];
  const float* wrow = wg + ((long)(b * NH + h) * NKP + k) * NEL;
  for (int e = 0; e < NEL; ++e) {
    int et = elim[b * NEL + e];
    acc += wrow[e] * xres[((long)b * NT + et) * CH + c] * ts[b * NT + et];
  }
  xk[((long)b * NKP + k) * CH + c] = acc / tsnew[b * NKP + k];
}

// ===========================================================================
extern "C" void kernel_launch(void* const* d_in, const int* in_sizes, int n_in,
                              void* d_out, int out_size, void* d_ws, size_t ws_size,
                              hipStream_t stream) {
  const float* x       = (const float*)d_in[0];   // [64,197,768]
  const float* ts      = (const float*)d_in[1];   // [64,197]
  const float* w_qkv   = (const float*)d_in[2];   // [2304,768]
  const float* w_proj  = (const float*)d_in[3];   // [768,768]
  const float* b_proj  = (const float*)d_in[4];
  const float* ln1_g   = (const float*)d_in[5];
  const float* ln1_b   = (const float*)d_in[6];
  const float* ln2_g   = (const float*)d_in[7];
  const float* ln2_b   = (const float*)d_in[8];
  const float* w_fc1   = (const float*)d_in[9];   // [3072,768]
  const float* b_fc1   = (const float*)d_in[10];
  const float* w_fc2   = (const float*)d_in[11];  // [768,3072]
  const float* b_fc2   = (const float*)d_in[12];
  float* outp = (float*)d_out;                    // x_out [64,99,768] ++ ts [64,99]

  // ---- workspace bump allocator (assumes ws_size ~>= 400 MB) ----
  char* base = (char*)d_ws;
  size_t off = 0;
  auto alloc = [&](size_t bytes) -> void* {
    void* p = base + off;
    off += (bytes + 255) & ~(size_t)255;
    return p;
  };
  ushort_t* xnb    = (ushort_t*)alloc((size_t)ROWS * CH * 2);        // LN1(x) bf16
  ushort_t* wqkvb  = (ushort_t*)alloc((size_t)3 * CH * CH * 2);
  ushort_t* wprojb = (ushort_t*)alloc((size_t)CH * CH * 2);
  ushort_t* wfc1b  = (ushort_t*)alloc((size_t)HID * CH * 2);
  ushort_t* wfc2b  = (ushort_t*)alloc((size_t)CH * HID * 2);
  ushort_t* qkvb   = (ushort_t*)alloc((size_t)ROWS * 3 * CH * 2);    // [B,N,2304] bf16
  float*    attn   = (float*)alloc((size_t)NB * NH * SA2 * 4);       // logits->softmax
  float*    avg    = (float*)alloc((size_t)NB * SA2 * 4);
  ushort_t* aob    = (ushort_t*)alloc((size_t)ROWS * CH * 2);        // attn@v bf16
  float*    xres   = (float*)alloc((size_t)ROWS * CH * 4);           // x + proj out
  float*    lts    = (float*)alloc((size_t)NB * NT * 4);             // log(token_scales)
  int*      kept   = (int*)alloc((size_t)NB * NKP * 4);
  int*      elim   = (int*)alloc((size_t)NB * NEL * 4);
  float*    wg     = (float*)alloc((size_t)NB * NH * NW * 4);        // sparsified w
  float*    xk     = (float*)alloc((size_t)RK * CH * 4);             // x_kept_final
  ushort_t* hnb    = (ushort_t*)alloc((size_t)RK * CH * 2);          // LN2 bf16
  ushort_t* h1b    = (ushort_t*)alloc((size_t)RK * HID * 2);         // GELU(fc1) bf16
  (void)ws_size; (void)in_sizes; (void)n_in; (void)out_size;

  // 1) weights -> bf16 ; log(ts)
  {
    long n;
    n = (long)3 * CH * CH; cvt_bf16_kernel<<<(n + 255) / 256, 256, 0, stream>>>(w_qkv, wqkvb, n);
    n = (long)CH * CH;     cvt_bf16_kernel<<<(n + 255) / 256, 256, 0, stream>>>(w_proj, wprojb, n);
    n = (long)HID * CH;    cvt_bf16_kernel<<<(n + 255) / 256, 256, 0, stream>>>(w_fc1, wfc1b, n);
    n = (long)CH * HID;    cvt_bf16_kernel<<<(n + 255) / 256, 256, 0, stream>>>(w_fc2, wfc2b, n);
    n = (long)NB * NT;     log_kernel<<<(n + 255) / 256, 256, 0, stream>>>(ts, lts, n);
  }
  // 2) LN1 -> bf16
  ln_bf16_kernel<<<ROWS, 256, 0, stream>>>(x, ln1_g, ln1_b, xnb, CH);
  // 3) QKV GEMM (WMMA 2x2): [12608,768] x [2304,768]^T -> bf16
  {
    long tiles = (long)(ROWS / 32) * (3 * CH / 32);
    gemm_wmma_kernel<0><<<(tiles + 3) / 4, 128, 0, stream>>>(
        xnb, wqkvb, nullptr, nullptr, nullptr, qkvb, ROWS, 3 * CH, CH);
  }
  // 4) logits (WMMA) + log(ts) bias
  logits_wmma_kernel<<<dim3((169 + 3) / 4, NH, NB), 128, 0, stream>>>(qkvb, lts, attn);
  // 5) softmax in place
  softmax_kernel<<<NB * NH * NT, 256, 0, stream>>>(attn);
  // 6) head average for pagerank
  attn_avg_kernel<<<NB * NT, 256, 0, stream>>>(attn, avg);
  // 7) attn @ v -> bf16 (attention rows staged in LDS)
  attn_v_kernel<<<NB * NT, CH, 0, stream>>>(attn, qkvb, aob);
  // 8) proj GEMM (WMMA 2x2) + bias + residual(x) -> xres f32
  {
    long tiles = (long)(ROWS / 32) * (CH / 32);
    gemm_wmma_kernel<1><<<(tiles + 3) / 4, 128, 0, stream>>>(
        aob, wprojb, b_proj, x, xres, nullptr, ROWS, CH, CH);
  }
  // 9) pagerank + select (196x196 matrix in LDS)
  {
    size_t shm = ((size_t)196 * 197 + 196 * 4 + 256) * 4 + 196 * 4;
    pagerank_select_kernel<<<NB, 256, shm, stream>>>(avg, kept, elim);
  }
  // 10) gather + threshold-sparsify w
  {
    size_t shm = (size_t)NW * 4 + 256 * 4;
    gather_threshold_kernel<<<dim3(NH, NB), 256, shm, stream>>>(attn, kept, elim, wg);
  }
  // 11) ts_new -> tail of d_out  (then read back in propagate)
  float* tsout = outp + (size_t)RK * CH;
  ts_new_kernel<<<NB, 128, 0, stream>>>(wg, ts, kept, elim, tsout);
  // 12) sparse propagation -> xk
  propagate_kernel<<<dim3(NKP, NB), CH, 0, stream>>>(xres, wg, ts, tsout, kept, elim, xk);
  // 13) LN2 -> bf16
  ln_bf16_kernel<<<RK, 256, 0, stream>>>(xk, ln2_g, ln2_b, hnb, CH);
  // 14) fc1 GEMM (WMMA 2x2) + bias + GELU -> bf16
  {
    long tiles = (long)(RK / 32) * (HID / 32);
    gemm_wmma_kernel<2><<<(tiles + 3) / 4, 128, 0, stream>>>(
        hnb, wfc1b, b_fc1, nullptr, nullptr, h1b, RK, HID, CH);
  }
  // 15) fc2 GEMM (WMMA 2x2) + bias + residual(xk) -> d_out f32
  {
    long tiles = (long)(RK / 32) * (CH / 32);
    gemm_wmma_kernel<1><<<(tiles + 3) / 4, 128, 0, stream>>>(
        h1b, wfc2b, b_fc2, xk, outp, nullptr, RK, CH, HID);
  }
}
